// GUNet_uw_54485955117463
// MI455X (gfx1250) — matrix operand order repaired and checked by reading, not compiled, not run on hardware
//
#include <hip/hip_runtime.h>
#include <hip/hip_bf16.h>
#include <math.h>

// GraphUNet forward for MI455X (gfx1250, wave32).
//  - augment (B@B, integer-valued)  -> v_wmma_f32_16x16x32_bf16 (exact for counts <=256),
//    LDS-staged transposed B panel shared by 4 waves, coalesced b128 global loads.
//  - GCN GEMMs (x@W, A_hat@xw)      -> v_wmma_f32_16x16x4_f32 (fp32 fidelity)
//  - top-k: single-block bitonic sort matching lax.top_k (desc value, asc index ties)

#define FD    256
#define NODES 2048
#define NEDGE 65536

typedef float  v2f  __attribute__((ext_vector_type(2)));
typedef float  v8f  __attribute__((ext_vector_type(8)));
typedef __bf16 bf16_t;
typedef __bf16 v16bf __attribute__((ext_vector_type(16)));

static __device__ __forceinline__ float wave_sum(float v) {
#pragma unroll
  for (int m = 16; m > 0; m >>= 1) v += __shfl_xor(v, m, 32);
  return v;
}

// ---------------- elementwise / setup kernels ----------------

__global__ void k_zero(float* __restrict__ p, int n) {
  int i = blockIdx.x * blockDim.x + threadIdx.x;
  if (i < n) p[i] = 0.0f;
}

__global__ void k_copy(const float* __restrict__ a, float* __restrict__ b, int n) {
  int i = blockIdx.x * blockDim.x + threadIdx.x;
  if (i < n) b[i] = a[i];
}

__global__ void k_scatter_edges(const int* __restrict__ ei, float* __restrict__ A) {
  int e = blockIdx.x * blockDim.x + threadIdx.x;
  if (e >= NEDGE) return;
  int src = ei[e], dst = ei[NEDGE + e];
  atomicAdd(&A[(size_t)dst * NODES + src], 1.0f);
}

// deg[i] = sum_j A_hat[i,j]; dinv = rsqrt(deg); addd[i] = 2 if A[i,i]==0 else 0
__global__ void k_deg(const float* __restrict__ A, float* __restrict__ dinv,
                      float* __restrict__ addd, int n) {
  int wave = blockIdx.x * (blockDim.x >> 5) + (threadIdx.x >> 5);
  int lane = threadIdx.x & 31;
  if (wave >= n) return;
  const float* row = A + (size_t)wave * n;
  float sum = 0.f;
  for (int j = lane; j < n; j += 32) sum += row[j];
  sum = wave_sum(sum);
  if (lane == 0) {
    float ad  = (row[wave] == 0.0f) ? 2.0f : 0.0f;
    float deg = sum + ad;
    dinv[wave] = (deg > 0.0f) ? (1.0f / sqrtf(deg)) : 0.0f;
    addd[wave] = ad;
  }
}

__global__ void k_scale_rows(const float* __restrict__ xw, const float* __restrict__ dinv,
                             float* __restrict__ out, int tot) {
  int i = blockIdx.x * blockDim.x + threadIdx.x;
  if (i < tot) out[i] = xw[i] * dinv[i >> 8];  // FD == 256
}

// out = dinv[i]*(y + addd[i]*xw2) + bias[f]  (optional relu)
__global__ void k_gcn_epi(const float* __restrict__ y, const float* __restrict__ xw2,
                          const float* __restrict__ dinv, const float* __restrict__ addd,
                          const float* __restrict__ bias, float* __restrict__ out,
                          int tot, int relu) {
  int idx = blockIdx.x * blockDim.x + threadIdx.x;
  if (idx >= tot) return;
  int i = idx >> 8, f = idx & (FD - 1);
  float v = dinv[i] * (y[idx] + addd[i] * xw2[idx]) + bias[f];
  out[idx] = relu ? fmaxf(v, 0.0f) : v;
}

__global__ void k_prep_aug(const float* __restrict__ A, bf16_t* __restrict__ Bb, int n) {
  int idx = blockIdx.x * blockDim.x + threadIdx.x;
  if (idx >= n * n) return;
  int i = idx / n, j = idx - i * n;
  float v = (i == j) ? 1.0f : A[idx];   // remove self loops, add w=1 self loops
  Bb[idx] = (bf16_t)v;                  // exact: small integers
}

__global__ void k_score(const float* __restrict__ x, const float* __restrict__ p,
                        float* __restrict__ score, int n) {
  int wave = blockIdx.x * (blockDim.x >> 5) + (threadIdx.x >> 5);
  int lane = threadIdx.x & 31;
  if (wave >= n) return;
  float dot = 0.f, pp = 0.f;
  for (int f = lane; f < FD; f += 32) {
    float pv = p[f];
    dot += x[(size_t)wave * FD + f] * pv;
    pp  += pv * pv;
  }
  dot = wave_sum(dot); pp = wave_sum(pp);
  if (lane == 0) score[wave] = tanhf(dot / sqrtf(pp));
}

// Single-block bitonic sort: descending score, ascending index on ties (== lax.top_k).
__global__ void k_topk(const float* __restrict__ score, int n, int k,
                       int* __restrict__ perm, float* __restrict__ vals) {
  __shared__ float s[2048];
  __shared__ int   id[2048];
  int t = threadIdx.x;
  for (int i = t; i < n; i += blockDim.x) { s[i] = score[i]; id[i] = i; }
  for (int size = 2; size <= n; size <<= 1) {
    for (int stride = size >> 1; stride > 0; stride >>= 1) {
      __syncthreads();
      for (int i = t; i < (n >> 1); i += blockDim.x) {
        int pos = 2 * i - (i & (stride - 1));
        int j = pos + stride;
        bool desc = ((pos & size) == 0);
        float a = s[pos], b = s[j];
        int ia = id[pos], ib = id[j];
        bool aFirst = (a > b) || (a == b && ia < ib);
        if (desc ? !aFirst : aFirst) { s[pos] = b; s[j] = a; id[pos] = ib; id[j] = ia; }
      }
    }
  }
  __syncthreads();
  for (int i = t; i < k; i += blockDim.x) { perm[i] = id[i]; vals[i] = s[i]; }
}

__global__ void k_gather_pool(const float* __restrict__ x, const int* __restrict__ perm,
                              const float* __restrict__ vals, float* __restrict__ out, int k) {
  int idx = blockIdx.x * blockDim.x + threadIdx.x;
  if (idx >= k * FD) return;
  int i = idx >> 8, f = idx & (FD - 1);
  out[idx] = x[(size_t)perm[i] * FD + f] * vals[i];
}

__global__ void k_gather_adj(const float* __restrict__ Aaug, const int* __restrict__ perm,
                             float* __restrict__ Aout, int k, int n) {
  int idx = blockIdx.x * blockDim.x + threadIdx.x;
  if (idx >= k * k) return;
  int i = idx / k, j = idx - i * k;
  Aout[idx] = Aaug[(size_t)perm[i] * n + perm[j]];
}

// tmp[perm[i], f] += xsmall[i, f]  (perm entries unique -> no atomics)
__global__ void k_scatter_rows(float* __restrict__ tmp, const int* __restrict__ perm,
                               const float* __restrict__ xs, int k) {
  int idx = blockIdx.x * blockDim.x + threadIdx.x;
  if (idx >= k * FD) return;
  int i = idx >> 8, f = idx & (FD - 1);
  tmp[(size_t)perm[i] * FD + f] += xs[idx];
}

__global__ void k_relu_bn(const float* __restrict__ xin, const float* __restrict__ g,
                          const float* __restrict__ b, const float* __restrict__ rm,
                          const float* __restrict__ rv, float* __restrict__ out, int tot) {
  int idx = blockIdx.x * blockDim.x + threadIdx.x;
  if (idx >= tot) return;
  int f = idx & (FD - 1);
  float v = fmaxf(xin[idx], 0.0f);
  out[idx] = (v - rm[f]) * (1.0f / sqrtf(rv[f] + 1e-5f)) * g[f] + b[f];
}

__global__ void k_linear(const float* __restrict__ h, const float* __restrict__ W,
                         const float* __restrict__ b, float* __restrict__ out, int n) {
  int wave = blockIdx.x * (blockDim.x >> 5) + (threadIdx.x >> 5);
  int lane = threadIdx.x & 31;
  if (wave >= n) return;
  float s0 = 0.f, s1 = 0.f;
  for (int f = lane; f < FD; f += 32) {
    float hv = h[(size_t)wave * FD + f];
    s0 += hv * W[2 * f];
    s1 += hv * W[2 * f + 1];
  }
  s0 = wave_sum(s0); s1 = wave_sum(s1);
  if (lane == 0) { out[2 * wave] = s0 + b[0]; out[2 * wave + 1] = s1 + b[1]; }
}

// ---------------- WMMA GEMMs ----------------

// C[M,N] = A[M,K] @ B[K,N], row-major fp32. One wave -> 16x64 tile via
// v_wmma_f32_16x16x4_f32; A fragment reused across 4 N-subtiles.
__global__ void k_gemm_f32(const float* __restrict__ A, const float* __restrict__ B,
                           float* __restrict__ C, int M, int N, int K) {
  int wave = blockIdx.x * (blockDim.x >> 5) + (threadIdx.x >> 5);
  int colGroups = N >> 6;
  int tm = (wave / colGroups) << 4;
  int tn = (wave % colGroups) << 6;
  int lane = threadIdx.x & 31;
  int half = lane >> 4, ml = lane & 15;
  v8f c0 = {}, c1 = {}, c2 = {}, c3 = {};
  const float* arow = A + (size_t)(tm + ml) * K;
  for (int k0 = 0; k0 < K; k0 += 4) {
    v2f a;
    a.x = arow[k0 + 2 * half];
    a.y = arow[k0 + 2 * half + 1];
    const float* b0 = B + (size_t)(k0 + 2 * half) * N + tn + ml;
    const float* b1 = b0 + N;
    v2f bA; bA.x = b0[0];  bA.y = b1[0];
    v2f bB; bB.x = b0[16]; bB.y = b1[16];
    v2f bC; bC.x = b0[32]; bC.y = b1[32];
    v2f bD; bD.x = b0[48]; bD.y = b1[48];
    c0 = __builtin_amdgcn_wmma_f32_16x16x4_f32(false, a, false, bA, (short)0, c0, false, false);
    c1 = __builtin_amdgcn_wmma_f32_16x16x4_f32(false, a, false, bB, (short)0, c1, false, false);
    c2 = __builtin_amdgcn_wmma_f32_16x16x4_f32(false, a, false, bC, (short)0, c2, false, false);
    c3 = __builtin_amdgcn_wmma_f32_16x16x4_f32(false, a, false, bD, (short)0, c3, false, false);
  }
#pragma unroll
  for (int v = 0; v < 8; ++v) {
    size_t r = (size_t)(tm + v + 8 * half) * N + tn + ml;
    C[r] = c0[v]; C[r + 16] = c1[v]; C[r + 32] = c2[v]; C[r + 48] = c3[v];
  }
}

// C[n,n] = Bb @ Bb (bf16 in, f32 out), zeroing the diagonal (augment epilogue).
// Workgroup = 4 waves computing a 64x64 tile. The shared B panel (64k x 64n)
// is staged TRANSPOSED in LDS with coalesced 128-bit global loads, so each
// fragment read is two contiguous 16B LDS loads per lane. A rows are private
// per wave and contiguous in global -> direct b128 loads.
#define KSTAGE 64
#define LDP    72   // padded k-pitch (144 B: spreads banks, keeps 16B alignment)

__global__ void k_gemm_bf16_aug(const bf16_t* __restrict__ Bb, float* __restrict__ C, int n) {
  __shared__ bf16_t ldsBT[64 * LDP];   // [col_rel 0..63][k_rel 0..KSTAGE-1]
  int tid  = threadIdx.x;
  int lane = tid & 31;
  int wav  = tid >> 5;                 // 0..3
  int colBlocks = n >> 6;
  int bm = blockIdx.x / colBlocks;
  int bn = blockIdx.x - bm * colBlocks;
  int tm = (bm << 6) + (wav << 4);
  int tn = bn << 6;
  int half = lane >> 4, ml = lane & 15;

  v8f c0 = {}, c1 = {}, c2 = {}, c3 = {};
  const bf16_t* arow = Bb + (size_t)(tm + ml) * n;

  for (int k0 = 0; k0 < n; k0 += KSTAGE) {
    __syncthreads();   // previous compute done reading LDS
    // Stage: 64 rows (k) x 64 cols (n) bf16, 512 x uint4 chunks, 4 per thread.
    for (int t = tid; t < 512; t += 128) {
      int r = t >> 3;                  // k_rel
      int q = t & 7;                   // 8-col chunk
      const bf16_t* src = Bb + (size_t)(k0 + r) * n + tn + (q << 3);
      uint4 v = *(const uint4*)src;
      if (k0 + KSTAGE < n)             // prefetch next panel (global_prefetch_b8)
        __builtin_prefetch(src + (size_t)KSTAGE * n, 0, 1);
      bf16_t tmp[8];
      *(uint4*)tmp = v;
#pragma unroll
      for (int i = 0; i < 8; ++i)
        ldsBT[((q << 3) + i) * LDP + r] = tmp[i];
    }
    __syncthreads();
#pragma unroll
    for (int ks = 0; ks < KSTAGE; ks += 32) {
      v16bf a;
      {
        bf16_t ta[16];
        *(uint4*)(ta)     = *(const uint4*)(arow + k0 + ks + 16 * half);
        *(uint4*)(ta + 8) = *(const uint4*)(arow + k0 + ks + 16 * half + 8);
#pragma unroll
        for (int i = 0; i < 16; ++i) a[i] = ta[i];
      }
      v16bf bv0, bv1, bv2, bv3;
      {
        bf16_t tb[16];
        const bf16_t* p;
        p = &ldsBT[(ml +  0) * LDP + ks + 16 * half];
        *(uint4*)(tb) = *(const uint4*)p; *(uint4*)(tb + 8) = *(const uint4*)(p + 8);
#pragma unroll
        for (int i = 0; i < 16; ++i) bv0[i] = tb[i];
        p = &ldsBT[(ml + 16) * LDP + ks + 16 * half];
        *(uint4*)(tb) = *(const uint4*)p; *(uint4*)(tb + 8) = *(const uint4*)(p + 8);
#pragma unroll
        for (int i = 0; i < 16; ++i) bv1[i] = tb[i];
        p = &ldsBT[(ml + 32) * LDP + ks + 16 * half];
        *(uint4*)(tb) = *(const uint4*)p; *(uint4*)(tb + 8) = *(const uint4*)(p + 8);
#pragma unroll
        for (int i = 0; i < 16; ++i) bv2[i] = tb[i];
        p = &ldsBT[(ml + 48) * LDP + ks + 16 * half];
        *(uint4*)(tb) = *(const uint4*)p; *(uint4*)(tb + 8) = *(const uint4*)(p + 8);
#pragma unroll
        for (int i = 0; i < 16; ++i) bv3[i] = tb[i];
      }
      c0 = __builtin_amdgcn_wmma_f32_16x16x32_bf16(false, a, false, bv0, (short)0, c0, false, false);
      c1 = __builtin_amdgcn_wmma_f32_16x16x32_bf16(false, a, false, bv1, (short)0, c1, false, false);
      c2 = __builtin_amdgcn_wmma_f32_16x16x32_bf16(false, a, false, bv2, (short)0, c2, false, false);
      c3 = __builtin_amdgcn_wmma_f32_16x16x32_bf16(false, a, false, bv3, (short)0, c3, false, false);
    }
  }
#pragma unroll
  for (int v = 0; v < 8; ++v) {
    int gr = tm + v + 8 * half;
    int gc = tn + ml;
    size_t r = (size_t)gr * n;
    C[r + gc]      = (gr == gc)      ? 0.0f : c0[v];
    C[r + gc + 16] = (gr == gc + 16) ? 0.0f : c1[v];
    C[r + gc + 32] = (gr == gc + 32) ? 0.0f : c2[v];
    C[r + gc + 48] = (gr == gc + 48) ? 0.0f : c3[v];
  }
}

// ---------------- host orchestration ----------------

extern "C" void kernel_launch(void* const* d_in, const int* in_sizes, int n_in,
                              void* d_out, int out_size, void* d_ws, size_t ws_size,
                              hipStream_t stream) {
  const float* x_in  = (const float*)d_in[0];
  const int*   ei    = (const int*)d_in[1];
  const float* u1_dW = (const float*)d_in[2];
  const float* u1_db = (const float*)d_in[3];
  const float* u1_pp = (const float*)d_in[4];
  const float* u1_uW = (const float*)d_in[5];
  const float* u1_ub = (const float*)d_in[6];
  const float* u2_dW = (const float*)d_in[7];
  const float* u2_db = (const float*)d_in[8];
  const float* u2_pp = (const float*)d_in[9];
  const float* u2_uW = (const float*)d_in[10];
  const float* u2_ub = (const float*)d_in[11];
  const float* bn1_g = (const float*)d_in[12];
  const float* bn1_b = (const float*)d_in[13];
  const float* bn1_rm = (const float*)d_in[14];
  const float* bn1_rv = (const float*)d_in[15];
  const float* bn2_g = (const float*)d_in[16];
  const float* bn2_b = (const float*)d_in[17];
  const float* bn2_rm = (const float*)d_in[18];
  const float* bn2_rv = (const float*)d_in[19];
  const float* lin_W = (const float*)d_in[20];
  const float* lin_b = (const float*)d_in[21];

  // workspace carve (~62 MB)
  size_t off = 0;
  auto alloc = [&](size_t bytes) -> char* {
    char* p = (char*)d_ws + off;
    off += (bytes + 255) & ~(size_t)255;
    return p;
  };
  float*  A0   = (float*)alloc((size_t)NODES * NODES * 4);
  float*  AUG  = (float*)alloc((size_t)NODES * NODES * 4);
  bf16_t* BB   = (bf16_t*)alloc((size_t)NODES * NODES * 2);
  float*  A1   = (float*)alloc((size_t)1024 * 1024 * 4);
  float*  A2   = (float*)alloc((size_t)512 * 512 * 4);
  float*  A3   = (float*)alloc((size_t)256 * 256 * 4);
  float*  XS0  = (float*)alloc((size_t)NODES * FD * 4);
  float*  XS1  = (float*)alloc((size_t)1024 * FD * 4);
  float*  XS2  = (float*)alloc((size_t)512 * FD * 4);
  float*  XA   = (float*)alloc((size_t)NODES * FD * 4);
  float*  XB   = (float*)alloc((size_t)NODES * FD * 4);
  float*  XW   = (float*)alloc((size_t)NODES * FD * 4);
  float*  XW2  = (float*)alloc((size_t)NODES * FD * 4);
  float*  Yb   = (float*)alloc((size_t)NODES * FD * 4);
  float*  OUT1 = (float*)alloc((size_t)NODES * FD * 4);
  float*  H1   = (float*)alloc((size_t)NODES * FD * 4);
  float*  DINV = (float*)alloc(NODES * 4);
  float*  ADDD = (float*)alloc(NODES * 4);
  float*  SCORE = (float*)alloc(NODES * 4);
  int*    PERM0 = (int*)alloc(1024 * 4);
  int*    PERM1 = (int*)alloc(512 * 4);
  int*    PERM2 = (int*)alloc(256 * 4);
  float*  VALS0 = (float*)alloc(1024 * 4);
  float*  VALS1 = (float*)alloc(512 * 4);
  float*  VALS2 = (float*)alloc(256 * 4);

  auto gemm_f32 = [&](const float* A, const float* B, float* C, int M, int N, int K) {
    int waves = (M / 16) * (N / 64);
    k_gemm_f32<<<waves / 4, 128, 0, stream>>>(A, B, C, M, N, K);
  };

  auto run_gcn = [&](const float* Ain, const float* xin, const float* W, const float* bias,
                     float* xout, int n, int relu) {
    int tot = n * FD;
    k_deg<<<(n + 7) / 8, 256, 0, stream>>>(Ain, DINV, ADDD, n);
    gemm_f32(xin, W, XW, n, FD, FD);                     // xw = x @ W
    k_scale_rows<<<(tot + 255) / 256, 256, 0, stream>>>(XW, DINV, XW2, tot);
    gemm_f32(Ain, XW2, Yb, n, FD, n);                    // y = A @ (dinv * xw)
    k_gcn_epi<<<(tot + 255) / 256, 256, 0, stream>>>(Yb, XW2, DINV, ADDD, bias, xout, tot, relu);
  };

  auto run_augment = [&](const float* Ain, int n) {
    k_prep_aug<<<(n * n + 255) / 256, 256, 0, stream>>>(Ain, BB, n);
    int blocks = (n / 64) * (n / 64);
    k_gemm_bf16_aug<<<blocks, 128, 0, stream>>>(BB, AUG, n);
  };

  auto run_pool = [&](const float* xin, const float* p, float* xout, float* Aout,
                      int* perm, float* vals, int n, int k) {
    k_score<<<(n + 7) / 8, 256, 0, stream>>>(xin, p, SCORE, n);
    k_topk<<<1, 1024, 0, stream>>>(SCORE, n, k, perm, vals);
    k_gather_pool<<<(k * FD + 255) / 256, 256, 0, stream>>>(xin, perm, vals, xout, k);
    k_gather_adj<<<(k * k + 255) / 256, 256, 0, stream>>>(AUG, perm, Aout, k, n);
  };

  auto run_unet = [&](const float* xin, const float* dW, const float* db, const float* pp,
                      const float* uW, const float* ub, float* xout) {
    int ns[4] = {2048, 1024, 512, 256};
    float* Alev[3] = {A0, A1, A2};
    float* Xlev[3] = {XS0, XS1, XS2};
    int*   perms[3] = {PERM0, PERM1, PERM2};
    float* valsA[3] = {VALS0, VALS1, VALS2};

    run_gcn(A0, xin, dW, db, XS0, 2048, 1);
    const float* curA = A0;
    const float* curX = XS0;
    for (int i = 1; i <= 3; ++i) {
      int n = ns[i - 1], k = ns[i];
      run_augment(curA, n);
      float* Adst = (i < 3) ? Alev[i] : A3;
      run_pool(curX, pp + (size_t)(i - 1) * FD, XA, Adst, perms[i - 1], valsA[i - 1], n, k);
      float* Xdst = (i < 3) ? Xlev[i] : XB;
      run_gcn(Adst, XA, dW + (size_t)i * FD * FD, db + (size_t)i * FD, Xdst, k, 1);
      curA = Adst; curX = Xdst;
    }
    const float* xup = curX;  // bottom features, 256 x FD (== XB)
    for (int i = 0; i < 3; ++i) {
      int j = 2 - i;
      int n = ns[j], k = ns[j + 1];
      int tot = n * FD;
      k_copy<<<(tot + 255) / 256, 256, 0, stream>>>(Xlev[j], XA, tot);                 // res
      k_scatter_rows<<<(k * FD + 255) / 256, 256, 0, stream>>>(XA, perms[j], xup, k);  // +up
      float* dst = (i < 2) ? XB : xout;
      run_gcn(Alev[j], XA, uW + (size_t)i * FD * FD, ub + (size_t)i * FD, dst, n, (i < 2) ? 1 : 0);
      xup = XB;
    }
  };

  // adjacency: A[dst, src] += 1
  k_zero<<<(NODES * NODES + 255) / 256, 256, 0, stream>>>(A0, NODES * NODES);
  k_scatter_edges<<<NEDGE / 256, 256, 0, stream>>>(ei, A0);

  int totN = NODES * FD;
  run_unet(x_in, u1_dW, u1_db, u1_pp, u1_uW, u1_ub, OUT1);
  k_relu_bn<<<(totN + 255) / 256, 256, 0, stream>>>(OUT1, bn1_g, bn1_b, bn1_rm, bn1_rv, H1, totN);
  run_unet(H1, u2_dW, u2_db, u2_pp, u2_uW, u2_ub, OUT1);
  k_relu_bn<<<(totN + 255) / 256, 256, 0, stream>>>(OUT1, bn2_g, bn2_b, bn2_rm, bn2_rv, H1, totN);
  k_linear<<<NODES / 8, 256, 0, stream>>>(H1, lin_W, lin_b, (float*)d_out, NODES);
}
